// MLPLocal_50929722196828
// MI455X (gfx1250) — matrix operand order repaired and checked by reading, not compiled, not run on hardware
//
#include <hip/hip_runtime.h>

// ---- WMMA vector types (CDNA5 / gfx1250, wave32) ----
typedef __attribute__((ext_vector_type(16))) __bf16 v16bf;
typedef __attribute__((ext_vector_type(8)))  __bf16 v8bf;
typedef __attribute__((ext_vector_type(8)))  float  v8f;

#define WMMA_BF16(A, B, C) \
  __builtin_amdgcn_wmma_f32_16x16x32_bf16(false, (A), false, (B), (short)0, (C), false, false)

// ---- Problem constants ----
#define BATCH 2048
#define NAG   64     // agents per batch
#define DD    256    // input dim
#define HHID  1024   // hidden dim
#define EE    512    // embedding dim
#define OUTW  768    // D + E

// ---- LDS layout (dynamic shared), all bf16 [64][264] planes ----
#define ROW_STRIDE 264                          // bf16 per row (pad for banks)
#define PLANE_BF16 (NAG * ROW_STRIDE)           // 16896 bf16 = 33792 B
#define SMEM_BYTES (4 * PLANE_BF16 * 2)         // Xhi,Xlo,Hhi,Hlo = 135168 B

// Load 16 consecutive bf16 (two b128 loads) into a v16bf fragment.
__device__ inline v16bf ld16(const __bf16* p) {
  v8bf a = *(const v8bf*)p;
  v8bf b = *(const v8bf*)(p + 8);
  v16bf r;
#pragma unroll
  for (int i = 0; i < 8; ++i) { r[i] = a[i]; r[i + 8] = b[i]; }
  return r;
}

// A-fragment (16x32 bf16): elems 0-7 at k0, elems 8-15 at k0+16 (per-row LDS).
__device__ inline v16bf ldA(const __bf16* p) {
  v8bf a = *(const v8bf*)p;
  v8bf b = *(const v8bf*)(p + 16);
  v16bf r;
#pragma unroll
  for (int i = 0; i < 8; ++i) { r[i] = a[i]; r[i + 8] = b[i]; }
  return r;
}

// ---- Preprocess: transpose weight [K][N] -> [N][K] and split fp32 -> bf16 hi/lo ----
__global__ void prep_split_transpose(const float* __restrict__ W,
                                     __bf16* __restrict__ Thi,
                                     __bf16* __restrict__ Tlo,
                                     int K, int N) {
  int idx = blockIdx.x * 256 + threadIdx.x;
  if (idx >= K * N) return;
  int k = idx / N;
  int n = idx - k * N;
  float v = W[idx];
  __bf16 h = (__bf16)v;
  Thi[(size_t)n * K + k] = h;
  Tlo[(size_t)n * K + k] = (__bf16)(v - (float)h);
}

// ---- Fused MLP + leave-one-out pooling: one workgroup per batch ----
__global__ __launch_bounds__(256)
void mlp_loo_kernel(const float* __restrict__ X,
                    const float* __restrict__ b1,
                    const float* __restrict__ b2,
                    const __bf16* __restrict__ w1t_hi, const __bf16* __restrict__ w1t_lo,
                    const __bf16* __restrict__ w2t_hi, const __bf16* __restrict__ w2t_lo,
                    float* __restrict__ out) {
  extern __shared__ __bf16 smem[];
  __bf16* Xhi = smem;                   // [64][264] bf16
  __bf16* Xlo = smem + PLANE_BF16;
  __bf16* Hhi = smem + 2 * PLANE_BF16;
  __bf16* Hlo = smem + 3 * PLANE_BF16;

  const int b     = blockIdx.x;
  const int tid   = threadIdx.x;
  const int wave  = tid >> 5;
  const int lane  = tid & 31;
  const int l15   = lane & 15;
  const int lhalf = lane >> 4;

  const float* Xb   = X   + (size_t)b * (NAG * DD);
  float*       outb = out + (size_t)b * (NAG * OUTW);

  // Stage X into LDS (pre-split hi/lo ONCE) and write the passthrough output.
  for (int i = tid; i < NAG * DD; i += 256) {
    int n = i >> 8;        // / 256
    int d = i & 255;
    float v = Xb[i];
    outb[n * OUTW + d] = v;
    __bf16 h = (__bf16)v;
    Xhi[n * ROW_STRIDE + d] = h;
    Xlo[n * ROW_STRIDE + d] = (__bf16)(v - (float)h);
  }

  // emb accumulators: this wave owns emb columns [wave*64, wave*64+64).
  v8f eacc[4][4];
#pragma unroll
  for (int mt = 0; mt < 4; ++mt)
#pragma unroll
    for (int nt = 0; nt < 4; ++nt)
#pragma unroll
      for (int q = 0; q < 8; ++q) eacc[mt][nt][q] = 0.f;

  __syncthreads();

  for (int r = 0; r < 4; ++r) {
    const int hc = r * 256;  // this round handles H columns [hc, hc+256)

    // ---------- GEMM1: this wave computes H columns [hc + wave*32, +32) ----------
#pragma unroll
    for (int t = 0; t < 2; ++t) {
      const int ncol = hc + wave * 32 + t * 16 + l15;  // this lane's H column
      v8f acc1[4];
#pragma unroll
      for (int mt = 0; mt < 4; ++mt)
#pragma unroll
        for (int q = 0; q < 8; ++q) acc1[mt][q] = 0.f;

      for (int kc = 0; kc < DD; kc += 32) {
        // B fragment (32x16): elems 0-7 at kc+lhalf*8, elems 8-15 at kc+16+lhalf*8 in W1T row.
        const __bf16* pbh = w1t_hi + (size_t)ncol * DD + kc + lhalf * 8;
        const __bf16* pbl = w1t_lo + (size_t)ncol * DD + kc + lhalf * 8;
        v16bf Bh = ldA(pbh);
        v16bf Bl = ldA(pbl);
#pragma unroll
        for (int mt = 0; mt < 4; ++mt) {
          const int aoff = (mt * 16 + l15) * ROW_STRIDE + kc + lhalf * 8;
          v16bf Ah = ldA(Xhi + aoff);
          v16bf Al = ldA(Xlo + aoff);
          acc1[mt] = WMMA_BF16(Ah, Bh, acc1[mt]);
          acc1[mt] = WMMA_BF16(Ah, Bl, acc1[mt]);
          acc1[mt] = WMMA_BF16(Al, Bh, acc1[mt]);
        }
      }

      // Epilogue: bias + relu, split to bf16 hi/lo, store round buffer in LDS.
      const float bias = b1[ncol];
      const int col_local = wave * 32 + t * 16 + l15;
#pragma unroll
      for (int mt = 0; mt < 4; ++mt) {
#pragma unroll
        for (int q = 0; q < 8; ++q) {
          float h = acc1[mt][q] + bias;
          h = h > 0.f ? h : 0.f;
          __bf16 hh = (__bf16)h;
          __bf16 hl = (__bf16)(h - (float)hh);
          int m = mt * 16 + lhalf * 8 + q;
          Hhi[m * ROW_STRIDE + col_local] = hh;
          Hlo[m * ROW_STRIDE + col_local] = hl;
        }
      }
    }
    __syncthreads();

    // ---------- GEMM2: all waves consume this round's 256 H columns ----------
    for (int kk = 0; kk < 256; kk += 32) {
      v16bf Ah[4], Al[4];
#pragma unroll
      for (int mt = 0; mt < 4; ++mt) {
        const int aoff = (mt * 16 + l15) * ROW_STRIDE + kk + lhalf * 8;
        Ah[mt] = ldA(Hhi + aoff);
        Al[mt] = ldA(Hlo + aoff);
      }
#pragma unroll
      for (int nt = 0; nt < 4; ++nt) {
        const int ncol = wave * 64 + nt * 16 + l15;
        const __bf16* pbh = w2t_hi + (size_t)ncol * HHID + hc + kk + lhalf * 8;
        const __bf16* pbl = w2t_lo + (size_t)ncol * HHID + hc + kk + lhalf * 8;
        v16bf Bh = ldA(pbh);
        v16bf Bl = ldA(pbl);
#pragma unroll
        for (int mt = 0; mt < 4; ++mt) {
          eacc[mt][nt] = WMMA_BF16(Ah[mt], Bh, eacc[mt][nt]);
          eacc[mt][nt] = WMMA_BF16(Ah[mt], Bl, eacc[mt][nt]);
          eacc[mt][nt] = WMMA_BF16(Al[mt], Bh, eacc[mt][nt]);
        }
      }
    }
    __syncthreads();
  }

  // ---------- Leave-one-out pooling epilogue ----------
  // C-layout: lane holds (m = mt*16 + lhalf*8 + q, n = l15). Sum over this wave's
  // 32 m values in-register, fold the other lane-half with a shuffle.
#pragma unroll
  for (int nt = 0; nt < 4; ++nt) {
    const int ncol = wave * 64 + nt * 16 + l15;
    const float bias = b2[ncol];
    float partial = 0.f;
#pragma unroll
    for (int mt = 0; mt < 4; ++mt)
#pragma unroll
      for (int q = 0; q < 8; ++q) partial += eacc[mt][nt][q] + bias;
    float total = partial + __shfl_xor(partial, 16, 32);

    float* o = outb + DD + ncol;
#pragma unroll
    for (int mt = 0; mt < 4; ++mt) {
#pragma unroll
      for (int q = 0; q < 8; ++q) {
        int m = mt * 16 + lhalf * 8 + q;
        float e = eacc[mt][nt][q] + bias;
        o[m * OUTW] = (total - e) * (1.0f / 63.0f);
      }
    }
  }
}

extern "C" void kernel_launch(void* const* d_in, const int* in_sizes, int n_in,
                              void* d_out, int out_size, void* d_ws, size_t ws_size,
                              hipStream_t stream) {
  const float* X  = (const float*)d_in[0];
  const float* W1 = (const float*)d_in[1];
  const float* b1 = (const float*)d_in[2];
  const float* W2 = (const float*)d_in[3];
  const float* b2 = (const float*)d_in[4];
  float* out = (float*)d_out;

  // Workspace: W1T hi/lo (512 KB each) + W2T hi/lo (1 MB each) = 3 MB.
  char* ws = (char*)d_ws;
  __bf16* w1t_hi = (__bf16*)(ws);
  __bf16* w1t_lo = (__bf16*)(ws + 512 * 1024);
  __bf16* w2t_hi = (__bf16*)(ws + 1024 * 1024);
  __bf16* w2t_lo = (__bf16*)(ws + 2 * 1024 * 1024);

  prep_split_transpose<<<(DD * HHID + 255) / 256, 256, 0, stream>>>(W1, w1t_hi, w1t_lo, DD, HHID);
  prep_split_transpose<<<(HHID * EE + 255) / 256, 256, 0, stream>>>(W2, w2t_hi, w2t_lo, HHID, EE);

  mlp_loo_kernel<<<BATCH, 256, SMEM_BYTES, stream>>>(
      X, b1, b2, w1t_hi, w1t_lo, w2t_hi, w2t_lo, out);
}
